// HunyuanVideoDoubleStreamBlock_72559177499128
// MI455X (gfx1250) — compile-verified, matrix operand order checked by reading, not run on hardware
//
#include <hip/hip_runtime.h>
#include <math.h>

// ---------------- problem constants ----------------
constexpr int SI = 1024;
constexpr int ST = 256;
constexpr int S_ALL = SI + ST;     // 1280
constexpr int C = 3072;
constexpr int H = 24;
constexpr int D = 128;
constexpr int MLP = 4 * C;         // 12288
constexpr int NC6 = 6 * C;         // 18432
constexpr int QKV = 3 * C;         // 9216
constexpr float EPS = 1e-6f;
constexpr float SCALE = 0.08838834764831845f; // 1/sqrt(128)

typedef __bf16 bf16;
typedef __attribute__((ext_vector_type(8)))  float v8f;
typedef __attribute__((ext_vector_type(8)))  bf16  v8bf;
typedef __attribute__((ext_vector_type(16))) bf16  v16bf;

// ---------------- CDNA5 async global->LDS copy (ASYNCcnt path) ----------------
// LDS byte address = low 32 bits of the generic shared pointer (flat->LDS
// aperture mapping truncates to addr[31:0]).
__device__ __forceinline__ void async_copy_b128(void* lds_dst, const void* gsrc) {
  unsigned int  la = (unsigned int)(unsigned long long)(uintptr_t)lds_dst;
  unsigned long long ga = (unsigned long long)(uintptr_t)gsrc;
  asm volatile("global_load_async_to_lds_b128 %0, %1, off"
               :: "v"(la), "v"(ga) : "memory");
}
__device__ __forceinline__ void wait_async0() {
  asm volatile("s_wait_asynccnt 0" ::: "memory");
}

// ---------------- WMMA fragment helpers ----------------
// CDNA5 16-bit A/B operand layout (ISA 7.12.2): lanes 0-15 hold rows, K in
// {0..7,16..23}; lanes 16-31 hold the same rows, K in {8..15,24..31}.
__device__ __forceinline__ v16bf ld_frag_ptr(const bf16* p) {
  union { v16bf v; v8bf h[2]; } f;
  f.h[0] = *(const v8bf*)(p);
  f.h[1] = *(const v8bf*)(p + 16);
  return f.v;
}

__device__ __forceinline__ v16bf frag_ab(const bf16* tile, int ld, int row0, int kk) {
  const int lane = threadIdx.x & 31;
  const int half = lane >> 4;
  const int r = lane & 15;
  return ld_frag_ptr(tile + (row0 + r) * ld + kk + half * 8);
}

// ---------------- kernel 1: silu(vec) @ mod_w + mod_b ----------------
__global__ __launch_bounds__(256)
void silu_mod_kernel(const float* __restrict__ vec, const float* __restrict__ w,
                     const float* __restrict__ b, float* __restrict__ out) {
  __shared__ float sv[C];
  const int tid = threadIdx.x;
  for (int k = tid; k < C; k += 256) {
    float x = vec[k];
    sv[k] = x / (1.0f + __expf(-x));
  }
  __syncthreads();
  const int j = blockIdx.x * 256 + tid;
  float acc = 0.0f;
  for (int k = 0; k < C; k += 4) {
    acc += sv[k + 0] * w[(size_t)(k + 0) * NC6 + j];
    acc += sv[k + 1] * w[(size_t)(k + 1) * NC6 + j];
    acc += sv[k + 2] * w[(size_t)(k + 2) * NC6 + j];
    acc += sv[k + 3] * w[(size_t)(k + 3) * NC6 + j];
  }
  out[j] = acc + b[j];
}

// ---------------- kernel 2: layernorm + modulation -> bf16 ----------------
__global__ __launch_bounds__(256)
void ln_mod_kernel(const float* __restrict__ x, const float* __restrict__ mod,
                   int shift_off, int scale_off, bf16* __restrict__ y) {
  __shared__ float rs[256];
  __shared__ float rq[256];
  const int row = blockIdx.x;
  const int tid = threadIdx.x;
  const float* xr = x + (size_t)row * C;
  float s = 0.0f, q = 0.0f;
  for (int k = tid; k < C; k += 256) { float v = xr[k]; s += v; q += v * v; }
  rs[tid] = s; rq[tid] = q;
  __syncthreads();
  for (int st = 128; st > 0; st >>= 1) {
    if (tid < st) { rs[tid] += rs[tid + st]; rq[tid] += rq[tid + st]; }
    __syncthreads();
  }
  const float mean = rs[0] * (1.0f / C);
  const float var  = rq[0] * (1.0f / C) - mean * mean;
  const float rstd = rsqrtf(var + EPS);
  for (int k = tid; k < C; k += 256) {
    float g = (xr[k] - mean) * rstd;
    y[(size_t)row * C + k] = (bf16)(g * (1.0f + mod[scale_off + k]) + mod[shift_off + k]);
  }
}

// ---------------- kernel 3: tiled WMMA GEMM, double-buffered ----------------
// out[M,N] = epi(A_bf16[M,K] @ W_f32[K,N] + bias)
#define BM 128
#define BN 128
#define BKG 64
#define LDT 72  // BKG + 8, keeps 16B alignment (72*2 = 144 = 9*16)
#define GEMM_TILE (BM * LDT)
#define GEMM_SMEM_BYTES (4 * GEMM_TILE * 2)  // 2x A + 2x B tiles, bf16

enum { EPI_F32 = 0, EPI_GELU_BF16 = 1, EPI_RESGATE = 2 };

template <int EPI>
__global__ __launch_bounds__(256)
void gemm_kernel(const bf16* __restrict__ A, const float* __restrict__ W,
                 const float* __restrict__ bias, void* __restrict__ outv,
                 const float* __restrict__ base, const float* __restrict__ gate,
                 int M, int N, int K) {
  extern __shared__ __align__(16) char gsm[];
  bf16* smem = (bf16*)gsm;   // [0,2*TILE): A buffers, [2*TILE,4*TILE): B buffers

  const int tid = threadIdx.x;
  const int m0 = blockIdx.y * BM;
  const int n0 = blockIdx.x * BN;
  const int wave = tid >> 5;
  const int wm = wave & 1;   // 2 waves along M  -> 64 rows each
  const int wn = wave >> 1;  // 4 waves along N  -> 32 cols each

  v8f acc[4][2] = {};

  // Issue all loads for one K-tile into buffer `buf` (buf is runtime-variant,
  // so the shared-pointer arithmetic stays dynamic — no constant addrspacecast).
  auto load_tile = [&](int buf, int k0) {
    bf16* at = smem + buf * GEMM_TILE;
    bf16* bt = smem + (2 + buf) * GEMM_TILE;
    // A tile: BM x BKG bf16 -> async DMA straight into LDS (no VGPR staging)
#pragma unroll
    for (int c = 0; c < 4; ++c) {
      int idx = tid + c * 256;          // 1024 chunks of 8 elems
      int row = idx >> 3;               // 8 chunks per row
      int col = (idx & 7) * 8;
      async_copy_b128(&at[row * LDT + col],
                      &A[(size_t)(m0 + row) * K + k0 + col]);
    }
    if (k0 + BKG < K)
      __builtin_prefetch(&W[(size_t)(k0 + BKG) * N + n0], 0, 0);
    // W tile: BKG x BN fp32 -> bf16 transposed into Bt[N][K] (needs VALU convert)
#pragma unroll
    for (int c = 0; c < 8; ++c) {
      int idx = tid + c * 256;          // float4 chunks, 32 per row
      int kr = idx >> 5;
      int nc = (idx & 31) * 4;
      const float4 f = *(const float4*)&W[(size_t)(k0 + kr) * N + n0 + nc];
      bt[(nc + 0) * LDT + kr] = (bf16)f.x;
      bt[(nc + 1) * LDT + kr] = (bf16)f.y;
      bt[(nc + 2) * LDT + kr] = (bf16)f.z;
      bt[(nc + 3) * LDT + kr] = (bf16)f.w;
    }
  };

  load_tile(0, 0);
  wait_async0();
  __syncthreads();

  int cur = 0;
  for (int k0 = 0; k0 < K; k0 += BKG) {
    // kick off next tile's fill while we compute on the current one
    if (k0 + BKG < K) load_tile(cur ^ 1, k0 + BKG);
    const bf16* At = smem + cur * GEMM_TILE;
    const bf16* Bt = smem + (2 + cur) * GEMM_TILE;
#pragma unroll
    for (int kk = 0; kk < BKG; kk += 32) {
      v16bf af[4], bfr[2];
#pragma unroll
      for (int i = 0; i < 4; ++i) af[i] = frag_ab(At, LDT, wm * 64 + i * 16, kk);
#pragma unroll
      for (int j = 0; j < 2; ++j) bfr[j] = frag_ab(Bt, LDT, wn * 32 + j * 16, kk);
#pragma unroll
      for (int i = 0; i < 4; ++i)
#pragma unroll
        for (int j = 0; j < 2; ++j)
          acc[i][j] = __builtin_amdgcn_wmma_f32_16x16x32_bf16(
              false, af[i], false, bfr[j], (short)0, acc[i][j], false, false);
    }
    wait_async0();
    __syncthreads();
    cur ^= 1;
  }

  // epilogue.  C/D layout: VGPR v, lanes 0-15 -> (M=v, N=lane); lanes 16-31 -> (M=v+8, N=lane-16)
  const int lane = tid & 31;
  const int half = lane >> 4;
  const int r = lane & 15;
#pragma unroll
  for (int i = 0; i < 4; ++i) {
#pragma unroll
    for (int j = 0; j < 2; ++j) {
      const int n = n0 + wn * 32 + j * 16 + r;
      const float bn = bias[n];
      const float gn = (EPI == EPI_RESGATE) ? gate[n] : 0.0f;
#pragma unroll
      for (int v = 0; v < 8; ++v) {
        const int m = m0 + wm * 64 + i * 16 + half * 8 + v;
        const float val = acc[i][j][v] + bn;
        const size_t o = (size_t)m * N + n;
        if (EPI == EPI_F32) {
          ((float*)outv)[o] = val;
        } else if (EPI == EPI_GELU_BF16) {
          const float u = val;
          const float g = 0.5f * u *
              (1.0f + tanhf(0.7978845608028654f * (u + 0.044715f * u * u * u)));
          ((bf16*)outv)[o] = (bf16)g;
        } else {
          ((float*)outv)[o] = base[o] + val * gn;
        }
      }
    }
  }
}

// ---------------- kernel 4: qk rms-norm + rope + reorder to [H][S][D] bf16 ----
__global__ __launch_bounds__(128)
void qkv_prep_kernel(const float* __restrict__ qkv_i, const float* __restrict__ qkv_t,
                     const float* __restrict__ i_qn, const float* __restrict__ i_kn,
                     const float* __restrict__ t_qn, const float* __restrict__ t_kn,
                     const float* __restrict__ fcos, const float* __restrict__ fsin,
                     bf16* __restrict__ Qo, bf16* __restrict__ Ko, bf16* __restrict__ Vo) {
  __shared__ float red[128];
  const int s = blockIdx.x;
  const int h = blockIdx.y;
  const int d = threadIdx.x;
  const bool img = (s < SI);
  const float* row = img ? (qkv_i + (size_t)s * QKV) : (qkv_t + (size_t)(s - SI) * QKV);
  float qv = row[(0 * H + h) * D + d];
  float kv = row[(1 * H + h) * D + d];
  float vv = row[(2 * H + h) * D + d];

  // sum(q^2)
  red[d] = qv * qv;  __syncthreads();
  for (int st = 64; st > 0; st >>= 1) { if (d < st) red[d] += red[d + st]; __syncthreads(); }
  const float q2 = red[0];  __syncthreads();
  // sum(k^2)
  red[d] = kv * kv;  __syncthreads();
  for (int st = 64; st > 0; st >>= 1) { if (d < st) red[d] += red[d + st]; __syncthreads(); }
  const float k2 = red[0];

  const float* qw = img ? i_qn : t_qn;
  const float* kw = img ? i_kn : t_kn;
  float qn = qv * rsqrtf(q2 * (1.0f / D) + EPS) * qw[d];
  float kn = kv * rsqrtf(k2 * (1.0f / D) + EPS) * kw[d];

  if (img) {
    const float cv = fcos[(size_t)s * D + d];
    const float sv = fsin[(size_t)s * D + d];
    const float qo = __shfl_xor(qn, 1);   // wave32 neighbor within rope pair
    const float ko = __shfl_xor(kn, 1);
    const float qr = (d & 1) ? qo : -qo;  // rot_half
    const float kr = (d & 1) ? ko : -ko;
    qn = qn * cv + qr * sv;
    kn = kn * cv + kr * sv;
  }
  const size_t o = ((size_t)h * S_ALL + s) * D + d;
  Qo[o] = (bf16)qn;
  Ko[o] = (bf16)kn;
  Vo[o] = (bf16)vv;
}

// ---------------- kernel 5: flash attention ----------------
#define LDSK 136  // 128 + 8 bf16, keeps 16B row alignment (136*2 = 272 = 17*16)
#define ATTN_SMEM_BYTES ((3 * 128 * LDSK + 8 * 16 * LDSK) * 2)

__global__ __launch_bounds__(256)
void attn_kernel(const bf16* __restrict__ Q, const bf16* __restrict__ Kb,
                 const bf16* __restrict__ Vb, bf16* __restrict__ O) {
  extern __shared__ __align__(16) char smem_raw[];
  bf16* Ks = (bf16*)smem_raw;          // [128][LDSK]  K chunk, key-major
  bf16* Vt = Ks + 128 * LDSK;          // [128][LDSK]  V chunk transposed: [d][key]
  bf16* Ps = Vt + 128 * LDSK;          // 8 waves x [16][LDSK]

  const int tid = threadIdx.x;
  const int lane = tid & 31;
  const int wave = tid >> 5;
  const int half = lane >> 4;
  const int r = lane & 15;
  const int h = blockIdx.y;
  const int qrow0 = blockIdx.x * 128 + wave * 16;

  const bf16* Qh = Q + (size_t)h * S_ALL * D;
  const bf16* Kh = Kb + (size_t)h * S_ALL * D;
  const bf16* Vh = Vb + (size_t)h * S_ALL * D;
  bf16* Pw = Ps + wave * 16 * LDSK;

  // Q fragments held in registers: 16 rows x 128 d = 4 A-frags
  v16bf qf[4];
#pragma unroll
  for (int kk = 0; kk < 4; ++kk)
    qf[kk] = ld_frag_ptr(Qh + (size_t)(qrow0 + r) * D + kk * 32 + half * 8);

  float mrow[8], lrow[8];
  v8f of[8] = {};
#pragma unroll
  for (int v = 0; v < 8; ++v) { mrow[v] = -1e30f; lrow[v] = 0.0f; }

  for (int c0 = 0; c0 < S_ALL; c0 += 128) {
    // K chunk: async DMA to LDS; V chunk: manual transpose through VGPRs
#pragma unroll
    for (int c = 0; c < 8; ++c) {
      int idx = tid + c * 256;          // 2048 chunks of 8 bf16
      int row = idx >> 4;               // 16 chunks per row
      int col = (idx & 15) * 8;
      async_copy_b128(&Ks[row * LDSK + col], &Kh[(size_t)(c0 + row) * D + col]);
      v8bf vv = *(const v8bf*)&Vh[(size_t)(c0 + row) * D + col];
#pragma unroll
      for (int e = 0; e < 8; ++e) Vt[(col + e) * LDSK + row] = vv[e];
    }
    wait_async0();
    __syncthreads();

    // S = Q @ K^T  (B^T == K chunk, key-major — load like an A frag)
    v8f sa[8] = {};
#pragma unroll
    for (int kk = 0; kk < 4; ++kk) {
#pragma unroll
      for (int j = 0; j < 8; ++j) {
        v16bf bfr = frag_ab(Ks, LDSK, j * 16, kk * 32);
        sa[j] = __builtin_amdgcn_wmma_f32_16x16x32_bf16(
            false, qf[kk], false, bfr, (short)0, sa[j], false, false);
      }
    }

    // scale + per-row max (row v of half -> reduce across 16 lanes)
    float rmax[8];
#pragma unroll
    for (int v = 0; v < 8; ++v) {
      float mx = -1e30f;
#pragma unroll
      for (int j = 0; j < 8; ++j) {
        float sc = sa[j][v] * SCALE;
        sa[j][v] = sc;
        mx = fmaxf(mx, sc);
      }
      mx = fmaxf(mx, __shfl_xor(mx, 1));
      mx = fmaxf(mx, __shfl_xor(mx, 2));
      mx = fmaxf(mx, __shfl_xor(mx, 4));
      mx = fmaxf(mx, __shfl_xor(mx, 8));
      rmax[v] = mx;
    }

    float corr[8];
#pragma unroll
    for (int v = 0; v < 8; ++v) {
      const float mn = fmaxf(mrow[v], rmax[v]);
      corr[v] = __expf(mrow[v] - mn);
      mrow[v] = mn;
    }

    // P = exp(S - m); row sums; restripe P into LDS for the PV matmul
#pragma unroll
    for (int v = 0; v < 8; ++v) {
      float rs = 0.0f;
#pragma unroll
      for (int j = 0; j < 8; ++j) {
        const float p = __expf(sa[j][v] - mrow[v]);
        rs += p;
        Pw[(v + half * 8) * LDSK + j * 16 + r] = (bf16)p;
      }
      rs += __shfl_xor(rs, 1);
      rs += __shfl_xor(rs, 2);
      rs += __shfl_xor(rs, 4);
      rs += __shfl_xor(rs, 8);
      lrow[v] = lrow[v] * corr[v] + rs;
#pragma unroll
      for (int f = 0; f < 8; ++f) of[f][v] = of[f][v] * corr[v];
    }
    asm volatile("s_wait_dscnt 0" ::: "memory");  // wave-private P: writes before frag reads

    // O += P @ V   (B^T == Vt: [d][key])
#pragma unroll
    for (int kk = 0; kk < 4; ++kk) {
      v16bf a = frag_ab(Pw, LDSK, 0, kk * 32);
#pragma unroll
      for (int f = 0; f < 8; ++f) {
        v16bf bfr = frag_ab(Vt, LDSK, f * 16, kk * 32);
        of[f] = __builtin_amdgcn_wmma_f32_16x16x32_bf16(
            false, a, false, bfr, (short)0, of[f], false, false);
      }
    }
    __syncthreads();
  }

  // normalize and write o[s][h*D+d] as bf16 (feeds the projection GEMM)
#pragma unroll
  for (int f = 0; f < 8; ++f) {
    const int d = f * 16 + r;
#pragma unroll
    for (int v = 0; v < 8; ++v) {
      const int s = qrow0 + half * 8 + v;
      O[(size_t)s * C + h * D + d] = (bf16)(of[f][v] / lrow[v]);
    }
  }
}

// ---------------- host orchestration ----------------
extern "C" void kernel_launch(void* const* d_in, const int* in_sizes, int n_in,
                              void* d_out, int out_size, void* d_ws, size_t ws_size,
                              hipStream_t stream) {
  (void)in_sizes; (void)n_in; (void)out_size; (void)ws_size;

  const float* img   = (const float*)d_in[0];
  const float* txt   = (const float*)d_in[1];
  const float* vec   = (const float*)d_in[2];
  const float* fcos  = (const float*)d_in[3];
  const float* fsin  = (const float*)d_in[4];
  const float* i_mod_w = (const float*)d_in[5];
  const float* i_mod_b = (const float*)d_in[6];
  const float* i_qkv_w = (const float*)d_in[7];
  const float* i_qkv_b = (const float*)d_in[8];
  const float* i_qn  = (const float*)d_in[9];
  const float* i_kn  = (const float*)d_in[10];
  const float* i_pw  = (const float*)d_in[11];
  const float* i_pb  = (const float*)d_in[12];
  const float* i_f1w = (const float*)d_in[13];
  const float* i_f1b = (const float*)d_in[14];
  const float* i_f2w = (const float*)d_in[15];
  const float* i_f2b = (const float*)d_in[16];
  const float* t_mod_w = (const float*)d_in[17];
  const float* t_mod_b = (const float*)d_in[18];
  const float* t_qkv_w = (const float*)d_in[19];
  const float* t_qkv_b = (const float*)d_in[20];
  const float* t_qn  = (const float*)d_in[21];
  const float* t_kn  = (const float*)d_in[22];
  const float* t_pw  = (const float*)d_in[23];
  const float* t_pb  = (const float*)d_in[24];
  const float* t_f1w = (const float*)d_in[25];
  const float* t_f1b = (const float*)d_in[26];
  const float* t_f2w = (const float*)d_in[27];
  const float* t_f2b = (const float*)d_in[28];

  float* out_img = (float*)d_out;
  float* out_txt = out_img + (size_t)SI * C;

  // workspace carve-up
  char* wp = (char*)d_ws;
  auto carve = [&](size_t bytes) -> char* {
    char* p = wp;
    wp += (bytes + 255) & ~(size_t)255;
    return p;
  };
  float* mod_i  = (float*)carve((size_t)NC6 * 4);
  float* mod_t  = (float*)carve((size_t)NC6 * 4);
  bf16*  xln_i  = (bf16*)carve((size_t)SI * C * 2);
  bf16*  xln_t  = (bf16*)carve((size_t)ST * C * 2);
  float* qkv_i  = (float*)carve((size_t)SI * QKV * 4);
  float* qkv_t  = (float*)carve((size_t)ST * QKV * 4);
  bf16*  qh     = (bf16*)carve((size_t)H * S_ALL * D * 2);
  bf16*  kh     = (bf16*)carve((size_t)H * S_ALL * D * 2);
  bf16*  vh     = (bf16*)carve((size_t)H * S_ALL * D * 2);
  bf16*  attn   = (bf16*)carve((size_t)S_ALL * C * 2);
  float* x2_i   = (float*)carve((size_t)SI * C * 4);
  float* x2_t   = (float*)carve((size_t)ST * C * 4);
  bf16*  mlp_i  = (bf16*)carve((size_t)SI * C * 2);
  bf16*  mlp_t  = (bf16*)carve((size_t)ST * C * 2);
  bf16*  hid_i  = (bf16*)carve((size_t)SI * MLP * 2);
  bf16*  hid_t  = (bf16*)carve((size_t)ST * MLP * 2);

  // 1) modulation vectors
  silu_mod_kernel<<<NC6 / 256, 256, 0, stream>>>(vec, i_mod_w, i_mod_b, mod_i);
  silu_mod_kernel<<<NC6 / 256, 256, 0, stream>>>(vec, t_mod_w, t_mod_b, mod_t);

  // 2) LN + mod (shift1 @ 0, scale1 @ C)
  ln_mod_kernel<<<SI, 256, 0, stream>>>(img, mod_i, 0, C, xln_i);
  ln_mod_kernel<<<ST, 256, 0, stream>>>(txt, mod_t, 0, C, xln_t);

  // 3) QKV projections
  gemm_kernel<EPI_F32><<<dim3(QKV / BN, SI / BM), 256, GEMM_SMEM_BYTES, stream>>>(
      xln_i, i_qkv_w, i_qkv_b, qkv_i, nullptr, nullptr, SI, QKV, C);
  gemm_kernel<EPI_F32><<<dim3(QKV / BN, ST / BM), 256, GEMM_SMEM_BYTES, stream>>>(
      xln_t, t_qkv_w, t_qkv_b, qkv_t, nullptr, nullptr, ST, QKV, C);

  // 4) q/k rms-norm + rope + head-major bf16 reorder
  qkv_prep_kernel<<<dim3(S_ALL, H), 128, 0, stream>>>(
      qkv_i, qkv_t, i_qn, i_kn, t_qn, t_kn, fcos, fsin, qh, kh, vh);

  // 5) flash attention
  attn_kernel<<<dim3(S_ALL / 128, H), 256, ATTN_SMEM_BYTES, stream>>>(qh, kh, vh, attn);

  // 6) output projections with gate g1 (offset 2C) + residual
  gemm_kernel<EPI_RESGATE><<<dim3(C / BN, SI / BM), 256, GEMM_SMEM_BYTES, stream>>>(
      attn, i_pw, i_pb, x2_i, img, mod_i + 2 * C, SI, C, C);
  gemm_kernel<EPI_RESGATE><<<dim3(C / BN, ST / BM), 256, GEMM_SMEM_BYTES, stream>>>(
      attn + (size_t)SI * C, t_pw, t_pb, x2_t, txt, mod_t + 2 * C, ST, C, C);

  // 7) second LN + mod (shift2 @ 3C, scale2 @ 4C)
  ln_mod_kernel<<<SI, 256, 0, stream>>>(x2_i, mod_i, 3 * C, 4 * C, mlp_i);
  ln_mod_kernel<<<ST, 256, 0, stream>>>(x2_t, mod_t, 3 * C, 4 * C, mlp_t);

  // 8) MLP up + gelu
  gemm_kernel<EPI_GELU_BF16><<<dim3(MLP / BN, SI / BM), 256, GEMM_SMEM_BYTES, stream>>>(
      mlp_i, i_f1w, i_f1b, hid_i, nullptr, nullptr, SI, MLP, C);
  gemm_kernel<EPI_GELU_BF16><<<dim3(MLP / BN, ST / BM), 256, GEMM_SMEM_BYTES, stream>>>(
      mlp_t, t_f1w, t_f1b, hid_t, nullptr, nullptr, ST, MLP, C);

  // 9) MLP down with gate g2 (offset 5C) + residual -> final outputs
  gemm_kernel<EPI_RESGATE><<<dim3(C / BN, SI / BM), 256, GEMM_SMEM_BYTES, stream>>>(
      hid_i, i_f2w, i_f2b, out_img, x2_i, mod_i + 5 * C, SI, C, MLP);
  gemm_kernel<EPI_RESGATE><<<dim3(C / BN, ST / BM), 256, GEMM_SMEM_BYTES, stream>>>(
      hid_t, t_f2w, t_f2b, out_txt, x2_t, mod_t + 5 * C, ST, C, MLP);
}
